// WindowAttention_53128745451952
// MI455X (gfx1250) — compile-verified, matrix operand order checked by reading, not run on hardware
//
#include <hip/hip_runtime.h>
#include <hip/hip_bf16.h>

typedef __attribute__((ext_vector_type(16))) _Float16 v16h;
typedef __attribute__((ext_vector_type(4)))  _Float16 v4h;
typedef __attribute__((ext_vector_type(8)))  float    v8f;

#define DIM   384
#define NHEAD 12
#define NTOK  49
#define NPAD  64
#define HD    32
#define NWIN  64

#define WMMA_F16(a, b, c) \
  __builtin_amdgcn_wmma_f32_16x16x32_f16(false, (a), false, (b), (short)0, (c), false, false)

__device__ inline v8f vzero8() {
  v8f z;
#pragma unroll
  for (int j = 0; j < 8; ++j) z[j] = 0.f;
  return z;
}

// WMMA A/B fragment from f16 global (pre-converted weights).
// Lane layout (ISA 7.12.2): lanes 0-15 hold K = {0..7, 16..23}, lanes 16-31
// hold K = {8..15, 24..31}; caller passes kbase = k0 + (half ? 8 : 0).
__device__ inline v16h frag_from_f16g(const _Float16* __restrict__ base, int row,
                                      int ld, int kbase) {
  v16h f;
  const _Float16* p = base + (size_t)row * ld + kbase;
#pragma unroll
  for (int j = 0; j < 8; ++j) {
    f[j]     = p[j];
    f[j + 8] = p[j + 16];
  }
  return f;
}

// Same fragment build from an f16 LDS stage (K-dim contiguous in memory).
__device__ inline v16h frag_from_lds(const _Float16* base, int row, int ld, int kbase) {
  v16h f;
  const _Float16* p = base + row * ld + kbase;
#pragma unroll
  for (int j = 0; j < 8; ++j) {
    f[j]     = p[j];
    f[j + 8] = p[j + 16];
  }
  return f;
}

// ---------- prep: convert qkv_w / proj_w to f16 once into d_ws ----------
__global__ __launch_bounds__(256) void cvt_weights(const float* __restrict__ qkv_w,
                                                   const float* __restrict__ proj_w,
                                                   _Float16* __restrict__ dst) {
  const int nq = 3 * DIM * DIM;   // 442368
  const int np = DIM * DIM;       // 147456
  const int i = blockIdx.x * blockDim.x + threadIdx.x;
  if (i < nq) dst[i] = (_Float16)qkv_w[i];
  if (i < np) dst[nq + i] = (_Float16)proj_w[i];
}

__global__ __launch_bounds__(128, 1) void swin_window_attn(
    const float* __restrict__ x, const float* __restrict__ maskp,
    const _Float16* __restrict__ qkv_wh, const float* __restrict__ qkv_b,
    const _Float16* __restrict__ proj_wh, const float* __restrict__ proj_b,
    const float* __restrict__ bias_table, const int* __restrict__ rel_idx,
    float* __restrict__ out) {
  __shared__ _Float16 x_lds[NPAD * DIM];    // 48 KB: window input, f16, zero-padded rows
  __shared__ _Float16 wscratch[4 * 6144];   // 48 KB: per-wave Q|K|Vt (P reuses Q+K)
  __shared__ _Float16 o_lds[NPAD * DIM];    // 48 KB: concat-head attention output

  const int b    = blockIdx.x;
  const int tid  = threadIdx.x;
  const int lane = tid & 31;
  const int wave = tid >> 5;
  const int l15  = lane & 15;
  const int half = lane >> 4;
  const int kb0  = half * 8;      // K sub-offset for this half-wave

  const float* xb    = x + (size_t)b * NTOK * DIM;
  const float* maskb = maskp + (size_t)(b & (NWIN - 1)) * NTOK * NTOK;

  // ---- stage x as f16 into LDS once (rows >= NTOK zeroed) ----
  for (int r = wave; r < NPAD; r += 4) {
    const float* src = xb + (size_t)r * DIM + lane * 12;
    _Float16* dst = x_lds + r * DIM + lane * 12;
    if (r < NTOK) {
#pragma unroll
      for (int j = 0; j < 12; ++j) dst[j] = (_Float16)src[j];
    } else {
#pragma unroll
      for (int j = 0; j < 12; ++j) dst[j] = (_Float16)0.f;
    }
  }
  __syncthreads();

  _Float16* Qs = wscratch + wave * 6144;
  _Float16* Ks = Qs + 2048;
  _Float16* Vt = Qs + 4096;
  _Float16* Ps = Qs;              // overlaps Q/K once their fragments are in VGPRs

  const float scale = 0.17677669529663687f;   // 32^-0.5, folded into Q

#pragma unroll 1
  for (int h = wave; h < NHEAD; h += 4) {
    // ---------------- QKV projection (3 passes: Q, K, V) ----------------
#pragma unroll 1
    for (int which = 0; which < 3; ++which) {
      const int wrow0 = which * DIM + h * HD;
      v8f acc[4][2];
#pragma unroll
      for (int mt = 0; mt < 4; ++mt)
#pragma unroll
        for (int dt = 0; dt < 2; ++dt) acc[mt][dt] = vzero8();

#pragma unroll 1
      for (int kt = 0; kt < 12; ++kt) {
        const int k0 = kt * 32 + kb0;
        v16h a[4];
#pragma unroll
        for (int mt = 0; mt < 4; ++mt)
          a[mt] = frag_from_lds(x_lds, mt * 16 + l15, DIM, k0);
#pragma unroll
        for (int dt = 0; dt < 2; ++dt) {
          v16h bf = frag_from_f16g(qkv_wh, wrow0 + dt * 16 + l15, DIM, k0);
#pragma unroll
          for (int mt = 0; mt < 4; ++mt)
            acc[mt][dt] = WMMA_F16(a[mt], bf, acc[mt][dt]);
        }
      }
#pragma unroll
      for (int dt = 0; dt < 2; ++dt) {
        const float bv = qkv_b[wrow0 + dt * 16 + l15];
#pragma unroll
        for (int mt = 0; mt < 4; ++mt)
#pragma unroll
          for (int v = 0; v < 8; ++v) {
            const int M = mt * 16 + v + half * 8;
            const int d = dt * 16 + l15;
            const float val = acc[mt][dt][v] + bv;
            if (which == 0)      Qs[M * HD + d] = (_Float16)(val * scale);
            else if (which == 1) Ks[M * HD + d] = (_Float16)val;
            else {
              // key-axis permutation m' = 4*(M&15) + (M>>4), matches P store
              const int mp = 4 * (v + half * 8) + mt;
              Vt[d * NPAD + mp] = (_Float16)val;
            }
          }
      }
    }

    // -------- S = Q @ K^T + bias + mask, softmax, P store (f16) --------
    {
      v16h aq[4], bk[4];
#pragma unroll
      for (int mt = 0; mt < 4; ++mt)
        aq[mt] = frag_from_lds(Qs, mt * 16 + l15, HD, kb0);
#pragma unroll
      for (int nt = 0; nt < 4; ++nt)
        bk[nt] = frag_from_lds(Ks, nt * 16 + l15, HD, kb0);

#pragma unroll 1
      for (int mt = 0; mt < 4; ++mt) {
        v8f srow[4];
#pragma unroll
        for (int nt = 0; nt < 4; ++nt)
          srow[nt] = WMMA_F16(aq[mt], bk[nt], vzero8());

#pragma unroll
        for (int v = 0; v < 8; ++v) {
          const int M = mt * 16 + v + half * 8;
          float vals[4];
          float rmax = -3.0e38f;
#pragma unroll
          for (int nt = 0; nt < 4; ++nt) {
            const int Nc = nt * 16 + l15;
            float sv = srow[nt][v];
            if (Nc < NTOK) {
              if (M < NTOK) {
                const int ri = rel_idx[M * NTOK + Nc];
                sv += bias_table[ri * NHEAD + h] + maskb[M * NTOK + Nc];
              }
            } else {
              sv = -3.0e38f;            // padded key column
            }
            vals[nt] = sv;
            rmax = fmaxf(rmax, sv);
          }
#pragma unroll
          for (int m = 8; m >= 1; m >>= 1)
            rmax = fmaxf(rmax, __shfl_xor(rmax, m, 32));   // within 16-lane half
          float rsum = 0.f;
#pragma unroll
          for (int nt = 0; nt < 4; ++nt) {
            vals[nt] = __expf(vals[nt] - rmax);
            rsum += vals[nt];
          }
#pragma unroll
          for (int m = 8; m >= 1; m >>= 1)
            rsum += __shfl_xor(rsum, m, 32);
          const float inv = __builtin_amdgcn_rcpf(rsum);   // v_rcp_f32
          // permuted packed store: lane's 4 probs are contiguous (one b64)
          v4h pq;
#pragma unroll
          for (int nt = 0; nt < 4; ++nt) pq[nt] = (_Float16)(vals[nt] * inv);
          *(v4h*)(Ps + M * NPAD + 4 * l15) = pq;
        }
      }
    }

    // ---------------- O = P @ V  (K-dim 64 -> 2 steps) -------------------
    // A (P) and B (Vt) both indexed along the permuted key axis: invariant.
    v8f o[4][2];
#pragma unroll
    for (int mt = 0; mt < 4; ++mt)
#pragma unroll
      for (int dt = 0; dt < 2; ++dt) o[mt][dt] = vzero8();

#pragma unroll
    for (int kt = 0; kt < 2; ++kt) {
      const int k0 = kt * 32 + kb0;
      v16h ap[4];
#pragma unroll
      for (int mt = 0; mt < 4; ++mt)
        ap[mt] = frag_from_lds(Ps, mt * 16 + l15, NPAD, k0);
#pragma unroll
      for (int dt = 0; dt < 2; ++dt) {
        v16h bv = frag_from_lds(Vt, dt * 16 + l15, NPAD, k0);
#pragma unroll
        for (int mt = 0; mt < 4; ++mt)
          o[mt][dt] = WMMA_F16(ap[mt], bv, o[mt][dt]);
      }
    }
#pragma unroll
    for (int mt = 0; mt < 4; ++mt)
#pragma unroll
      for (int dt = 0; dt < 2; ++dt)
#pragma unroll
        for (int v = 0; v < 8; ++v) {
          const int M = mt * 16 + v + half * 8;
          o_lds[M * DIM + h * HD + dt * 16 + l15] = (_Float16)o[mt][dt][v];
        }
  } // head loop

  __syncthreads();

  // ---------------- output projection: out = O @ proj_w^T + proj_b ------
#pragma unroll 1
  for (int ci = 0; ci < 6; ++ci) {
    const int c0 = (wave * 6 + ci) * 16;
    v8f po[4];
#pragma unroll
    for (int mt = 0; mt < 4; ++mt) po[mt] = vzero8();

#pragma unroll 1
    for (int kt = 0; kt < 12; ++kt) {
      const int k0 = kt * 32 + kb0;
      v16h a[4];
#pragma unroll
      for (int mt = 0; mt < 4; ++mt)
        a[mt] = frag_from_lds(o_lds, mt * 16 + l15, DIM, k0);
      v16h bw = frag_from_f16g(proj_wh, c0 + l15, DIM, k0);
#pragma unroll
      for (int mt = 0; mt < 4; ++mt)
        po[mt] = WMMA_F16(a[mt], bw, po[mt]);
    }
    const float pb = proj_b[c0 + l15];
#pragma unroll
    for (int mt = 0; mt < 4; ++mt)
#pragma unroll
      for (int v = 0; v < 8; ++v) {
        const int M = mt * 16 + v + half * 8;
        if (M < NTOK)
          out[(size_t)b * NTOK * DIM + (size_t)M * DIM + c0 + l15] = po[mt][v] + pb;
      }
  }
}

extern "C" void kernel_launch(void* const* d_in, const int* in_sizes, int n_in,
                              void* d_out, int out_size, void* d_ws, size_t ws_size,
                              hipStream_t stream) {
  const float* x      = (const float*)d_in[0];
  const float* mask   = (const float*)d_in[1];
  const float* qkv_w  = (const float*)d_in[2];
  const float* qkv_b  = (const float*)d_in[3];
  const float* proj_w = (const float*)d_in[4];
  const float* proj_b = (const float*)d_in[5];
  const float* btab   = (const float*)d_in[6];
  const int*   ridx   = (const int*)d_in[7];
  float* out = (float*)d_out;

  _Float16* w16 = (_Float16*)d_ws;                 // 3*384*384 + 384*384 f16 (~1.13 MB)
  const _Float16* qkv_wh  = w16;
  const _Float16* proj_wh = w16 + 3 * DIM * DIM;

  const int ncvt = 3 * DIM * DIM;                  // covers proj too (it's smaller)
  hipLaunchKernelGGL(cvt_weights, dim3((ncvt + 255) / 256), dim3(256), 0, stream,
                     qkv_w, proj_w, w16);

  const int Bw = in_sizes[0] / (NTOK * DIM);       // 4096 windows
  hipLaunchKernelGGL(swin_window_attn, dim3(Bw), dim3(128), 0, stream,
                     x, mask, qkv_wh, qkv_b, proj_wh, proj_b, btab, ridx, out);
}
